// SdcnTs_83090437308951
// MI455X (gfx1250) — matrix operand order repaired and checked by reading, not compiled, not run on hardware
//
#include <hip/hip_runtime.h>
#include <hip/hip_bf16.h>

#define NN 100000
#define NE 3200000
#define DD 256
#define MT (NN / 16)   // 6250 row tiles, exact

typedef __attribute__((ext_vector_type(16))) __bf16 v16bf;
typedef __attribute__((ext_vector_type(8)))  float  v8f;

__device__ __forceinline__ __bf16 f2bf(float f) {
    unsigned u = __builtin_bit_cast(unsigned, f);
    u += 0x7fffu + ((u >> 16) & 1u);                 // round-to-nearest-even
    unsigned short h = (unsigned short)(u >> 16);
    return __builtin_bit_cast(__bf16, h);
}

// ---------------------------------------------------------------------------
// Pack W (K x ncols, row-major f32) into WMMA B-fragment layout (bf16):
// packed[((ntile*8 + kk)*32 + lane)*16 + j]  with lane = column (lane&15),
// K = kk*32 + (lane>>4)*16 + j   (contiguous k order, same as the A fragment).
// Columns beyond ncols are zero-padded (used for the 4-wide W5).
// ---------------------------------------------------------------------------
__global__ void pack_w(const float* __restrict__ W, int ncols, int ntiles,
                       __bf16* __restrict__ out) {
    int idx = blockIdx.x * blockDim.x + threadIdx.x;
    if (idx >= ntiles * 256) return;
    int lane  = idx & 31;
    int kk    = (idx >> 5) & 7;
    int ntile = idx >> 8;
    int n     = ntile * 16 + (lane & 15);
    int kb    = kk * 32 + (lane >> 4) * 16;
    __bf16* dst = out + (size_t)idx * 16;
#pragma unroll
    for (int j = 0; j < 16; ++j) {
        float v = (n < ncols) ? W[(size_t)(kb + j) * ncols + n] : 0.0f;
        dst[j] = f2bf(v);
    }
}

// ---------------------------------------------------------------------------
// Dense GEMM: Y[M x (NT*16 per wave)] = blend(X) @ W,  K = 256 (8 steps of 32)
// blend(X) = (1-sigma)*maybe_relu(X) + sigma*T, converted to bf16 inline.
// One wave computes a 16 x (16*NT) strip with v_wmma_f32_16x16x32_bf16.
// blockDim = 32*WAVES; waves per block cover ntiles [wave*NT, wave*NT+NT).
// ---------------------------------------------------------------------------
template <int NT>
__global__ void gemm_bf16_wmma(const float* __restrict__ X, const float* __restrict__ T,
                               float sigma, int relu_x,
                               const __bf16* __restrict__ Wp,
                               float* __restrict__ Y, int ldy) {
    const int lane  = threadIdx.x & 31;
    const int wave  = threadIdx.x >> 5;
    const int mtile = blockIdx.x;
    const int nbase = wave * NT;

    const int hlf = lane >> 4;           // which 16-lane half
    const int m   = lane & 15;           // A row within tile
    const size_t rowoff = (size_t)(mtile * 16 + m) * DD;
    const float* __restrict__ xr = X + rowoff;
    const float* __restrict__ tr = T + rowoff;
    const float oms = 1.0f - sigma;

    v8f acc[NT];
#pragma unroll
    for (int t = 0; t < NT; ++t)
#pragma unroll
        for (int r = 0; r < 8; ++r) acc[t][r] = 0.0f;

#pragma unroll
    for (int kk = 0; kk < 8; ++kk) {
        // A fragment: 16x32 bf16; lane hlf offsets K by 16 (contiguous, matches packed B)
        v16bf a;
        const int kbase = kk * 32 + 16 * hlf;
#pragma unroll
        for (int j = 0; j < 8; ++j) {
            const int koff = kbase + 2 * j;
            float2 xv = *(const float2*)(xr + koff);
            if (relu_x) { xv.x = fmaxf(xv.x, 0.0f); xv.y = fmaxf(xv.y, 0.0f); }
            float2 tv = *(const float2*)(tr + koff);
            a[2 * j    ] = f2bf(oms * xv.x + sigma * tv.x);
            a[2 * j + 1] = f2bf(oms * xv.y + sigma * tv.y);
        }
#pragma unroll
        for (int t = 0; t < NT; ++t) {
            const v16bf b = *(const v16bf*)(Wp +
                ((size_t)((nbase + t) * 8 + kk) * 32 + lane) * 16);
            acc[t] = __builtin_amdgcn_wmma_f32_16x16x32_bf16(
                false, a, false, b, (short)0, acc[t], false, false);
        }
    }

    // C/D layout: VGPR r -> row r + 8*hlf, column = lane&15
    const int n = lane & 15;
#pragma unroll
    for (int t = 0; t < NT; ++t) {
        float* __restrict__ yb =
            Y + (size_t)(mtile * 16 + 8 * hlf) * ldy + (nbase + t) * 16 + n;
#pragma unroll
        for (int r = 0; r < 8; ++r) yb[(size_t)r * ldy] = acc[t][r];
    }
}

// ---------------------------------------------------------------------------
// SPMM (256 cols): one wave per edge; 2 x float4 gather per lane from Y[col],
// f32 atomic scatter-add into S[row]. Both matrices are L2-resident (102 MB).
// ---------------------------------------------------------------------------
__global__ void spmm256(const int* __restrict__ row, const int* __restrict__ col,
                        const float* __restrict__ val,
                        const float* __restrict__ Y, float* __restrict__ S) {
    const int lane = threadIdx.x & 31;
    int w        = (blockIdx.x * blockDim.x + threadIdx.x) >> 5;
    const int nw = (gridDim.x * blockDim.x) >> 5;
    for (int e = w; e < NE; e += nw) {
        const int   r = row[e];
        const int   c = col[e];
        const float v = val[e];
        if (e + nw < NE) {
            int cn = col[e + nw];
            __builtin_prefetch(Y + (size_t)cn * DD + lane * 4, 0, 3);
        }
        const float4* __restrict__ src = (const float4*)(Y + (size_t)c * DD) + lane;
        float* __restrict__ dst = S + (size_t)r * DD + lane * 4;
        float4 y0 = src[0];
        float4 y1 = src[32];
        atomicAdd(dst + 0,   v * y0.x);
        atomicAdd(dst + 1,   v * y0.y);
        atomicAdd(dst + 2,   v * y0.z);
        atomicAdd(dst + 3,   v * y0.w);
        atomicAdd(dst + 128, v * y1.x);
        atomicAdd(dst + 129, v * y1.y);
        atomicAdd(dst + 130, v * y1.z);
        atomicAdd(dst + 131, v * y1.w);
    }
}

// SPMM for layer 5 (4 valid cols, Y5 row stride 16): one thread per edge.
__global__ void spmm4(const int* __restrict__ row, const int* __restrict__ col,
                      const float* __restrict__ val,
                      const float* __restrict__ Y5, float* __restrict__ S5) {
    int i = blockIdx.x * blockDim.x + threadIdx.x;
    int n = gridDim.x * blockDim.x;
    for (int e = i; e < NE; e += n) {
        int   r = row[e], c = col[e];
        float v = val[e];
        float4 y = *(const float4*)(Y5 + (size_t)c * 16);
        atomicAdd(S5 + (size_t)r * 4 + 0, v * y.x);
        atomicAdd(S5 + (size_t)r * 4 + 1, v * y.y);
        atomicAdd(S5 + (size_t)r * 4 + 2, v * y.z);
        atomicAdd(S5 + (size_t)r * 4 + 3, v * y.w);
    }
}

__global__ void zero_f4(float4* __restrict__ p, int n4) {
    int i = blockIdx.x * blockDim.x + threadIdx.x;
    int n = gridDim.x * blockDim.x;
    for (; i < n4; i += n) p[i] = make_float4(0.f, 0.f, 0.f, 0.f);
}

// ---------------------------------------------------------------------------
// Epilogue: one wave per node. softmax(S5 row) -> out[0..3];
// student-t q (V=1 => q = 1/(1+dist2), exponent 1) in full f32 -> out[4..7].
// Cluster matrix is 4 KB (L0-resident); wave32 shuffle reductions.
// ---------------------------------------------------------------------------
__global__ void finalize_k(const float* __restrict__ S5, const float* __restrict__ z,
                           const float* __restrict__ cl, float* __restrict__ out) {
    const int lane = threadIdx.x & 31;
    int w        = (blockIdx.x * blockDim.x + threadIdx.x) >> 5;
    const int nw = (gridDim.x * blockDim.x) >> 5;
    const float4* __restrict__ c0 = (const float4*)cl;
    const float4* __restrict__ c1 = (const float4*)(cl + DD);
    const float4* __restrict__ c2 = (const float4*)(cl + 2 * DD);
    const float4* __restrict__ c3 = (const float4*)(cl + 3 * DD);
    for (int nidx = w; nidx < NN; nidx += nw) {
        const float4* __restrict__ zr = (const float4*)(z + (size_t)nidx * DD);
        float zz = 0.f, d0 = 0.f, d1 = 0.f, d2 = 0.f, d3 = 0.f;
        float s0 = 0.f, s1 = 0.f, s2 = 0.f, s3 = 0.f;
        for (int j = lane; j < DD / 4; j += 32) {
            float4 zv = zr[j];
            float4 a = c0[j], b = c1[j], c = c2[j], d = c3[j];
            zz += zv.x * zv.x + zv.y * zv.y + zv.z * zv.z + zv.w * zv.w;
            d0 += zv.x * a.x + zv.y * a.y + zv.z * a.z + zv.w * a.w;
            d1 += zv.x * b.x + zv.y * b.y + zv.z * b.z + zv.w * b.w;
            d2 += zv.x * c.x + zv.y * c.y + zv.z * c.z + zv.w * c.w;
            d3 += zv.x * d.x + zv.y * d.y + zv.z * d.z + zv.w * d.w;
            s0 += a.x * a.x + a.y * a.y + a.z * a.z + a.w * a.w;
            s1 += b.x * b.x + b.y * b.y + b.z * b.z + b.w * b.w;
            s2 += c.x * c.x + c.y * c.y + c.z * c.z + c.w * c.w;
            s3 += d.x * d.x + d.y * d.y + d.z * d.z + d.w * d.w;
        }
#pragma unroll
        for (int off = 16; off > 0; off >>= 1) {
            zz += __shfl_xor(zz, off);
            d0 += __shfl_xor(d0, off); d1 += __shfl_xor(d1, off);
            d2 += __shfl_xor(d2, off); d3 += __shfl_xor(d3, off);
            s0 += __shfl_xor(s0, off); s1 += __shfl_xor(s1, off);
            s2 += __shfl_xor(s2, off); s3 += __shfl_xor(s3, off);
        }
        if (lane == 0) {
            const float* p = S5 + (size_t)nidx * 4;
            float p0 = p[0], p1 = p[1], p2 = p[2], p3 = p[3];
            float mx = fmaxf(fmaxf(p0, p1), fmaxf(p2, p3));
            float e0 = __expf(p0 - mx), e1 = __expf(p1 - mx);
            float e2 = __expf(p2 - mx), e3 = __expf(p3 - mx);
            float es = e0 + e1 + e2 + e3;
            float* o = out + (size_t)nidx * 8;
            o[0] = e0 / es; o[1] = e1 / es; o[2] = e2 / es; o[3] = e3 / es;
            float q0 = 1.f / (1.f + (zz - 2.f * d0 + s0));
            float q1 = 1.f / (1.f + (zz - 2.f * d1 + s1));
            float q2 = 1.f / (1.f + (zz - 2.f * d2 + s2));
            float q3 = 1.f / (1.f + (zz - 2.f * d3 + s3));
            float qs = q0 + q1 + q2 + q3;
            o[4] = q0 / qs; o[5] = q1 / qs; o[6] = q2 / qs; o[7] = q3 / qs;
        }
    }
}

// ---------------------------------------------------------------------------
extern "C" void kernel_launch(void* const* d_in, const int* in_sizes, int n_in,
                              void* d_out, int out_size, void* d_ws, size_t ws_size,
                              hipStream_t stream) {
    (void)in_sizes; (void)n_in; (void)out_size; (void)ws_size;
    const float* enc   = (const float*)d_in[0];
    const float* tra1  = (const float*)d_in[1];
    const float* tra2  = (const float*)d_in[2];
    const float* tra3  = (const float*)d_in[3];
    const float* zin   = (const float*)d_in[4];
    const int*   erow  = (const int*)d_in[5];
    const int*   ecol  = (const int*)d_in[6];
    const float* evals = (const float*)d_in[7];
    const float* W1 = (const float*)d_in[8];
    const float* W2 = (const float*)d_in[9];
    const float* W3 = (const float*)d_in[10];
    const float* W4 = (const float*)d_in[11];
    const float* W5 = (const float*)d_in[12];
    const float* clus = (const float*)d_in[13];
    float* out = (float*)d_out;

    char* base = (char*)d_ws;
    size_t off = 0;
    auto take = [&](size_t bytes) -> void* {
        void* p = base + off;
        off += (bytes + 255) & ~(size_t)255;
        return p;
    };
    float*  SBUF = (float*)take((size_t)NN * DD * 4);   // 102.4 MB
    float*  YBUF = (float*)take((size_t)NN * DD * 4);   // 102.4 MB
    float*  Y5   = (float*)take((size_t)NN * 16 * 4);   // 6.4 MB (padded to 16 cols)
    float*  S5   = (float*)take((size_t)NN * 4 * 4);    // 1.6 MB
    __bf16* Wp1  = (__bf16*)take(16 * 8 * 32 * 16 * 2);
    __bf16* Wp2  = (__bf16*)take(16 * 8 * 32 * 16 * 2);
    __bf16* Wp3  = (__bf16*)take(16 * 8 * 32 * 16 * 2);
    __bf16* Wp4  = (__bf16*)take(16 * 8 * 32 * 16 * 2);
    __bf16* Wp5  = (__bf16*)take(1 * 8 * 32 * 16 * 2);

    // Pack weights into WMMA B-fragment layout (bf16)
    pack_w<<<16, 256, 0, stream>>>(W1, DD, 16, Wp1);
    pack_w<<<16, 256, 0, stream>>>(W2, DD, 16, Wp2);
    pack_w<<<16, 256, 0, stream>>>(W3, DD, 16, Wp3);
    pack_w<<<16, 256, 0, stream>>>(W4, DD, 16, Wp4);
    pack_w<<<1, 256, 0, stream>>>(W5, 4, 1, Wp5);

    const dim3 gG(MT), bG(128);       // 4 waves x 4 ntiles = 16 ntiles (N=256)
    // Layer 1: A = enc (no blend, no relu)
    gemm_bf16_wmma<4><<<gG, bG, 0, stream>>>(enc, enc, 0.0f, 0, Wp1, YBUF, DD);
    zero_f4<<<4096, 256, 0, stream>>>((float4*)SBUF, NN * DD / 4);
    spmm256<<<12288, 256, 0, stream>>>(erow, ecol, evals, YBUF, SBUF);
    // Layers 2..4: A = (1-s)*relu(S) + s*tra_i
    gemm_bf16_wmma<4><<<gG, bG, 0, stream>>>(SBUF, tra1, 0.5f, 1, Wp2, YBUF, DD);
    zero_f4<<<4096, 256, 0, stream>>>((float4*)SBUF, NN * DD / 4);
    spmm256<<<12288, 256, 0, stream>>>(erow, ecol, evals, YBUF, SBUF);
    gemm_bf16_wmma<4><<<gG, bG, 0, stream>>>(SBUF, tra2, 0.5f, 1, Wp3, YBUF, DD);
    zero_f4<<<4096, 256, 0, stream>>>((float4*)SBUF, NN * DD / 4);
    spmm256<<<12288, 256, 0, stream>>>(erow, ecol, evals, YBUF, SBUF);
    gemm_bf16_wmma<4><<<gG, bG, 0, stream>>>(SBUF, tra3, 0.5f, 1, Wp4, YBUF, DD);
    zero_f4<<<4096, 256, 0, stream>>>((float4*)SBUF, NN * DD / 4);
    spmm256<<<12288, 256, 0, stream>>>(erow, ecol, evals, YBUF, SBUF);
    // Layer 5 (N padded to 16): A = (1-s)*relu(S4) + s*z
    gemm_bf16_wmma<1><<<dim3(MT), dim3(32), 0, stream>>>(SBUF, zin, 0.5f, 1, Wp5, Y5, 16);
    zero_f4<<<1024, 256, 0, stream>>>((float4*)S5, NN);
    spmm4<<<8192, 256, 0, stream>>>(erow, ecol, evals, Y5, S5);
    // softmax + student-t q -> out[N, 8]
    finalize_k<<<3125, 256, 0, stream>>>(S5, zin, clus, out);
}
